// CustomLSTM_29686813950556
// MI455X (gfx1250) — matrix operand order repaired and checked by reading, not compiled, not run on hardware
//
#include <hip/hip_runtime.h>
#include <hip/hip_bf16.h>

// ---------------------------------------------------------------------------
// 2-layer LSTM for MI455X (gfx1250): bf16 WMMA (v_wmma_f32_16x16x32_bf16)
// with f32 accumulation. Input projection hoisted out of the time loop as a
// chunked large GEMM; recurrent GEMM fused with the gate nonlinearities.
// Tile staging is synchronous global->VGPR->LDS: the compiler software-
// pipelines it across the WMMA stream (verified in round-1 disasm); the
// asm-based async-LDS variant regressed codegen, so async copies are kept
// only in the cold c0-state copy.
// ---------------------------------------------------------------------------

typedef __attribute__((ext_vector_type(16))) __bf16 v16bf;
typedef __attribute__((ext_vector_type(8)))  __bf16 v8bf;
typedef __attribute__((ext_vector_type(8)))  float  v8f;

union BF16x16 { v16bf v; v8bf h[2]; };

constexpr int NB = 64;     // batch
constexpr int NT = 512;    // time
constexpr int ND = 1024;   // input dim
constexpr int NH = 1024;   // hidden dim
constexpr int NG = 4 * NH; // 4096 gate columns
constexpr int NK = 1024;   // K for both GEMMs
constexpr int CHUNK = 64;  // timesteps precomputed per x-proj GEMM
constexpr int MC = CHUNK * NB; // 4096 rows per chunk GEMM

// --- gfx1250 async global->LDS helpers (ASYNCcnt path, cold code only) -----
__device__ __forceinline__ unsigned lds_off(const void* p) {
    // generic pointer to a __shared__ object: addr[31:0] is the LDS offset
    return (unsigned)(size_t)p;
}
__device__ __forceinline__ void async_b128(unsigned lds, const void* g) {
    asm volatile("global_load_async_to_lds_b128 %0, %1, off"
                 :: "v"(lds), "v"((unsigned long long)(size_t)g) : "memory");
}

// ---------------------------------------------------------------------------
// x [B,T,D] f32  ->  xbf [T,B,D] bf16  (both sides coalesced over d)
// ---------------------------------------------------------------------------
__global__ __launch_bounds__(256) void convert_x_kernel(
    const float* __restrict__ x, __bf16* __restrict__ xbf)
{
    size_t i4 = ((size_t)blockIdx.x * blockDim.x + threadIdx.x) * 4;
    if (i4 >= (size_t)NB * NT * ND) return;
    int d  = (int)(i4 % ND);
    size_t bt = i4 / ND;
    int b = (int)(bt / NT);
    int t = (int)(bt % NT);
    const float4 v = *(const float4*)(x + i4);
    __bf16* dst = xbf + ((size_t)t * NB + b) * ND + d;
    dst[0] = (__bf16)v.x; dst[1] = (__bf16)v.y;
    dst[2] = (__bf16)v.z; dst[3] = (__bf16)v.w;
}

// ---------------------------------------------------------------------------
// W [K,N] f32 -> Wt [N,K] bf16 via LDS tile (so WMMA B-fragments read
// two 16-byte K-contiguous runs per lane).
// ---------------------------------------------------------------------------
__global__ __launch_bounds__(256) void transpose_w_kernel(
    const float* __restrict__ W, __bf16* __restrict__ Wt, int K, int N)
{
    __shared__ float tile[32][33];
    const int tx = threadIdx.x & 31, ty = threadIdx.x >> 5; // 32 x 8
    const int kb = blockIdx.x * 32, nb = blockIdx.y * 32;
    #pragma unroll
    for (int i = 0; i < 32; i += 8)
        tile[ty + i][tx] = W[(size_t)(kb + ty + i) * N + nb + tx];
    __syncthreads();
    #pragma unroll
    for (int i = 0; i < 32; i += 8)
        Wt[(size_t)(nb + ty + i) * K + kb + tx] = (__bf16)tile[tx][ty + i];
}

__global__ __launch_bounds__(256) void init_hbf_kernel(
    const float* __restrict__ h0, __bf16* __restrict__ hbf)
{
    int i = blockIdx.x * blockDim.x + threadIdx.x;
    if (i < NB * NH) hbf[i] = (__bf16)h0[i];
}

// ---------------------------------------------------------------------------
// c0 [2*NB*NH] f32 -> cs state slots in d_out, staged through LDS with the
// gfx1250 async global->LDS engine (each thread owns a 16B LDS slot, so its
// own s_wait_asynccnt 0 is sufficient ordering before the ds_load).
// ---------------------------------------------------------------------------
__global__ __launch_bounds__(256) void copy_c0_async_kernel(
    const float* __restrict__ src, float* __restrict__ dst)
{
    __shared__ alignas(16) float buf[256 * 4];
    const int tid = threadIdx.x;
    size_t off = ((size_t)blockIdx.x * 256 + tid) * 4;
    async_b128(lds_off(&buf[tid * 4]), src + off);
    asm volatile("s_wait_asynccnt 0" ::: "memory");
    float4 v = *(const float4*)&buf[tid * 4];
    *(float4*)(dst + off) = v;
}

// ---------------------------------------------------------------------------
// Chunk GEMM: C[4096,4096] = A[4096,1024](bf16) @ Bt[4096,1024]^T, f32 accum.
// 128x128 tile per workgroup, 8 waves of 64x32, LDS-staged, K step 32.
// ---------------------------------------------------------------------------
__global__ __launch_bounds__(256) void wmma_gemm_kernel(
    const __bf16* __restrict__ A,   // [MC, NK] row-major (chunk slice)
    const __bf16* __restrict__ Bt,  // [NG, NK] row-major (= W^T)
    float* __restrict__ C)          // [MC, NG] row-major
{
    __shared__ alignas(16) __bf16 As[128 * 32];
    __shared__ alignas(16) __bf16 Bs[128 * 32];

    const int tid  = threadIdx.x;
    const int wave = tid >> 5;
    const int lane = tid & 31;
    const int hf   = lane >> 4;       // lane-half selector
    const int l15  = lane & 15;
    const int mw   = (wave >> 2) * 64;  // wave M base in tile
    const int nw   = (wave & 3) * 32;   // wave N base in tile
    const int rowBlk = blockIdx.y * 128;
    const int colBlk = blockIdx.x * 128;

    v8f acc[4][2] = {};

    for (int kk = 0; kk < NK; kk += 32) {
        { // stage A tile 128x32 (8KB): 32B per thread
            int r = tid >> 1, seg = tid & 1;
            const int4* src = (const int4*)(A + (size_t)(rowBlk + r) * NK + kk + seg * 16);
            int4* dst = (int4*)(As + r * 32 + seg * 16);
            dst[0] = src[0]; dst[1] = src[1];
        }
        { // stage B tile 128x32 (8KB)
            int n = tid >> 1, seg = tid & 1;
            const int4* src = (const int4*)(Bt + (size_t)(colBlk + n) * NK + kk + seg * 16);
            int4* dst = (int4*)(Bs + n * 32 + seg * 16);
            dst[0] = src[0]; dst[1] = src[1];
        }
        if (kk + 32 < NK) { // prefetch next K slice into near cache
            __builtin_prefetch(A  + (size_t)(rowBlk + (tid >> 1)) * NK + kk + 32, 0, 3);
            __builtin_prefetch(Bt + (size_t)(colBlk + (tid >> 1)) * NK + kk + 32, 0, 3);
        }
        __syncthreads();

        BF16x16 a[4], b[2];
        #pragma unroll
        for (int i = 0; i < 4; i++) {   // A frag: halves at K+0 / K+16, +8 per lane-half
            const __bf16* base = As + (mw + i * 16 + l15) * 32;
            a[i].h[0] = *(const v8bf*)(base + hf * 8);
            a[i].h[1] = *(const v8bf*)(base + 16 + hf * 8);
        }
        #pragma unroll
        for (int j = 0; j < 2; j++) {   // B frag: halves at K+0 / K+16, +8 per lane-half
            const __bf16* base = Bs + (nw + j * 16 + l15) * 32;
            b[j].h[0] = *(const v8bf*)(base + hf * 8);
            b[j].h[1] = *(const v8bf*)(base + 16 + hf * 8);
        }
        #pragma unroll
        for (int i = 0; i < 4; i++)
            #pragma unroll
            for (int j = 0; j < 2; j++)
                acc[i][j] = __builtin_amdgcn_wmma_f32_16x16x32_bf16(
                    false, a[i].v, false, b[j].v, (short)0, acc[i][j], false, false);
        __syncthreads();
    }

    #pragma unroll
    for (int i = 0; i < 4; i++)
        #pragma unroll
        for (int j = 0; j < 2; j++) {
            int row0 = rowBlk + mw + i * 16 + hf * 8; // C/D: lane-half selects M+8
            int col  = colBlk + nw + j * 16 + l15;
            #pragma unroll
            for (int rr = 0; rr < 8; rr++)
                C[(size_t)(row0 + rr) * NG + col] = acc[i][j][rr];
        }
}

// ---------------------------------------------------------------------------
// One LSTM timestep: g = Gt + h@Wh^T + b ; gates ; c/h update.
// Each workgroup: 64 rows x 32 hidden cols across all 4 gate quadrants
// (= 64x128 GEMM tile). 8 waves, 1 N-tile x 4 M-tiles each.
// ---------------------------------------------------------------------------
__global__ __launch_bounds__(256) void lstm_step_kernel(
    const __bf16* __restrict__ hbf_in,  // [NB, NH]
    const __bf16* __restrict__ Wht,     // [NG, NH] (= Wh^T)
    const float*  __restrict__ Gt,      // [NB, NG] precomputed x-projection for t
    const float*  __restrict__ bias,    // [NG]
    float*        __restrict__ cstate,  // [NB, NH] in-place cell state
    float*        __restrict__ hA, size_t hA_rowstride, // primary f32 h out
    float*        __restrict__ hB,      // secondary f32 h out (nullable)
    __bf16*       __restrict__ hbf_out, // [NB, NH] next-step GEMM input
    __bf16*       __restrict__ ysbf)    // layer-0 sequence out (nullable)
{
    __shared__ alignas(16) __bf16 As[64 * 32];
    __shared__ alignas(16) __bf16 Bs[128 * 32];
    __shared__ float gs[64 * 128];

    const int tid  = threadIdx.x;
    const int wave = tid >> 5;
    const int lane = tid & 31;
    const int hf   = lane >> 4;
    const int l15  = lane & 15;
    const int jb   = blockIdx.x * 32;

    v8f acc[4] = {};

    for (int kk = 0; kk < NH; kk += 32) {
        { // stage A 64x32 (4KB): 16B per thread
            int r = tid >> 2, seg = tid & 3;
            *(int4*)(As + r * 32 + seg * 8) =
                *(const int4*)(hbf_in + (size_t)r * NH + kk + seg * 8);
        }
        { // stage B 128x32 (8KB): rows = 4 gate quadrants x 32 cols
            int n = tid >> 1, seg = tid & 1;
            int q = n >> 5, j = n & 31;
            const int4* src = (const int4*)(Wht + (size_t)(q * NH + jb + j) * NH + kk + seg * 16);
            int4* dst = (int4*)(Bs + n * 32 + seg * 16);
            dst[0] = src[0]; dst[1] = src[1];
        }
        __syncthreads();

        BF16x16 a[4], b;
        #pragma unroll
        for (int i = 0; i < 4; i++) {
            const __bf16* base = As + (i * 16 + l15) * 32;
            a[i].h[0] = *(const v8bf*)(base + hf * 8);
            a[i].h[1] = *(const v8bf*)(base + 16 + hf * 8);
        }
        {
            const __bf16* base = Bs + (wave * 16 + l15) * 32;
            b.h[0] = *(const v8bf*)(base + hf * 8);
            b.h[1] = *(const v8bf*)(base + 16 + hf * 8);
        }
        #pragma unroll
        for (int i = 0; i < 4; i++)
            acc[i] = __builtin_amdgcn_wmma_f32_16x16x32_bf16(
                false, a[i].v, false, b.v, (short)0, acc[i], false, false);
        __syncthreads();
    }

    // scatter accumulators to LDS so each thread can see all 4 gate quadrants
    #pragma unroll
    for (int i = 0; i < 4; i++) {
        int row0 = i * 16 + hf * 8;
        int col  = wave * 16 + l15;
        #pragma unroll
        for (int rr = 0; rr < 8; rr++)
            gs[(row0 + rr) * 128 + col] = acc[i][rr];
    }
    __syncthreads();

    // pointwise gates: 64 rows x 32 cols = 2048 elems, 8 per thread
    #pragma unroll
    for (int e = 0; e < 8; e++) {
        int idx = tid * 8 + e;
        int r = idx >> 5, j = idx & 31;
        int col = jb + j;
        float gi = gs[r * 128 +   0 + j] + Gt[(size_t)r * NG +          col] + bias[col];
        float gf = gs[r * 128 +  32 + j] + Gt[(size_t)r * NG + NH     + col] + bias[NH + col];
        float gn = gs[r * 128 +  64 + j] + Gt[(size_t)r * NG + 2 * NH + col] + bias[2 * NH + col];
        float go = gs[r * 128 +  96 + j] + Gt[(size_t)r * NG + 3 * NH + col] + bias[3 * NH + col];
        float it = 1.0f / (1.0f + __expf(-gi));
        float ft = 1.0f / (1.0f + __expf(-gf));
        float nt = tanhf(gn);
        float ot = 1.0f / (1.0f + __expf(-go));
        float c  = ft * cstate[(size_t)r * NH + col] + it * nt;
        float h  = ot * tanhf(c);
        cstate[(size_t)r * NH + col] = c;
        hA[(size_t)r * hA_rowstride + col] = h;
        if (hB) hB[(size_t)r * NH + col] = h;
        __bf16 hb = (__bf16)h;
        hbf_out[(size_t)r * NH + col] = hb;
        if (ysbf) ysbf[(size_t)r * NH + col] = hb;
    }
}

// ---------------------------------------------------------------------------
// Host orchestration
// ---------------------------------------------------------------------------
extern "C" void kernel_launch(void* const* d_in, const int* in_sizes, int n_in,
                              void* d_out, int out_size, void* d_ws, size_t ws_size,
                              hipStream_t stream) {
    const float* x   = (const float*)d_in[0];
    const float* h0  = (const float*)d_in[1];
    const float* c0  = (const float*)d_in[2];
    const float* Wx0 = (const float*)d_in[3];
    const float* Wh0 = (const float*)d_in[4];
    const float* b0  = (const float*)d_in[5];
    const float* Wx1 = (const float*)d_in[6];
    const float* Wh1 = (const float*)d_in[7];
    const float* b1  = (const float*)d_in[8];

    float* out      = (float*)d_out;                       // [B,T,H]
    float* out_hs   = out + (size_t)NB * NT * NH;          // [L,B,H]
    float* out_cs   = out_hs + (size_t)2 * NB * NH;        // [L,B,H]

    char* ws = (char*)d_ws;
    size_t o = 0;
    auto take = [&](size_t bytes) { char* p = ws + o; o += (bytes + 255) & ~(size_t)255; return p; };
    __bf16* xbf    = (__bf16*)take((size_t)NT * NB * ND * 2); // [T,B,D]
    __bf16* ys0bf  = (__bf16*)take((size_t)NT * NB * NH * 2); // [T,B,H]
    __bf16* Wx0t   = (__bf16*)take((size_t)NG * NK * 2);
    __bf16* Wh0t   = (__bf16*)take((size_t)NG * NK * 2);
    __bf16* Wx1t   = (__bf16*)take((size_t)NG * NK * 2);
    __bf16* Wh1t   = (__bf16*)take((size_t)NG * NK * 2);
    float*  Gchunk = (float*)take((size_t)MC * NG * 4);       // [CHUNK*B, 4H]
    __bf16* hbfA   = (__bf16*)take((size_t)NB * NH * 2);
    __bf16* hbfB   = (__bf16*)take((size_t)NB * NH * 2);

    // --- precision conversion / weight transpose / c-state init ---
    {
        size_t n4 = (size_t)NB * NT * ND / 4;
        convert_x_kernel<<<(unsigned)((n4 + 255) / 256), 256, 0, stream>>>(x, xbf);
        dim3 tg(NK / 32, NG / 32);
        transpose_w_kernel<<<tg, 256, 0, stream>>>(Wx0, Wx0t, NK, NG);
        transpose_w_kernel<<<tg, 256, 0, stream>>>(Wh0, Wh0t, NK, NG);
        transpose_w_kernel<<<tg, 256, 0, stream>>>(Wx1, Wx1t, NK, NG);
        transpose_w_kernel<<<tg, 256, 0, stream>>>(Wh1, Wh1t, NK, NG);
        // both layers' c0 -> cs state slots (async global->LDS engine)
        copy_c0_async_kernel<<<(2 * NB * NH) / (256 * 4), 256, 0, stream>>>(c0, out_cs);
    }

    for (int l = 0; l < 2; l++) {
        const __bf16* Asrc = (l == 0) ? xbf : ys0bf;
        const __bf16* Wxt  = (l == 0) ? Wx0t : Wx1t;
        const __bf16* Wht  = (l == 0) ? Wh0t : Wh1t;
        const float*  bias = (l == 0) ? b0 : b1;
        float*        cst  = out_cs + (size_t)l * NB * NH;

        init_hbf_kernel<<<(NB * NH + 255) / 256, 256, 0, stream>>>(h0 + (size_t)l * NB * NH, hbfA);

        for (int chunk = 0; chunk < NT / CHUNK; chunk++) {
            // input projection for CHUNK timesteps: [4096,1024] @ [1024,4096]
            wmma_gemm_kernel<<<dim3(NG / 128, MC / 128), 256, 0, stream>>>(
                Asrc + (size_t)chunk * MC * NK, Wxt, Gchunk);

            for (int tl = 0; tl < CHUNK; tl++) {
                int t = chunk * CHUNK + tl;
                const __bf16* hin  = (t & 1) ? hbfB : hbfA;
                __bf16*       hout = (t & 1) ? hbfA : hbfB;

                float* hA; size_t strideA; float* hB; __bf16* ys;
                if (l == 0) {
                    hA = out_hs;           strideA = NH;            // running h (final kept)
                    hB = nullptr;          ys = ys0bf + (size_t)t * NB * NH;
                } else {
                    hA = out + (size_t)t * NH; strideA = (size_t)NT * NH; // outputs[b,t,:]
                    hB = out_hs + (size_t)NB * NH; ys = nullptr;
                }
                lstm_step_kernel<<<NH / 32, 256, 0, stream>>>(
                    hin, Wht, Gchunk + (size_t)tl * NB * NG, bias,
                    cst, hA, strideA, hB, hout, ys);
            }
        }
    }
}
